// SharedMoEBlock_34548716929039
// MI455X (gfx1250) — matrix operand order repaired and checked by reading, not compiled, not run on hardware
//
#include <hip/hip_runtime.h>
#include <hip/hip_bf16.h>

// ---------------- problem constants ----------------
constexpr int kD = 1024, kH = 4096, kO = 1024, kE = 8, kT = 4096; // T = B*S
constexpr int BM = 128, BN = 128, BK = 32;     // block tile: 128x128, K-step 32
constexpr int TPE = kT / BM;                   // max row-tiles per expert = 32
constexpr int PAD_ROWS = 9216;                 // cap for 128-padded gathered rows

// ---------------- workspace layout (bytes) ----------------
constexpr size_t OFF_NORMED = 0;                                    // bf16 [kT][kD]
constexpr size_t OFF_HSH    = OFF_NORMED + (size_t)kT * kD * 2;     // bf16 [kT][kH]
constexpr size_t OFF_EH     = OFF_HSH + (size_t)kT * kH * 2;        // bf16 [PAD_ROWS][kH]
constexpr size_t OFF_TSW1   = OFF_EH + (size_t)PAD_ROWS * kH * 2;   // bf16 [kH][kD]  (sw1^T)
constexpr size_t OFF_TSW2   = OFF_TSW1 + (size_t)kD * kH * 2;       // bf16 [kO][kH]  (sw2^T)
constexpr size_t OFF_TEW1   = OFF_TSW2 + (size_t)kH * kO * 2;       // bf16 [kE][kH][kD]
constexpr size_t OFF_TEW2   = OFF_TEW1 + (size_t)kE * kD * kH * 2;  // bf16 [kE][kO][kH]
constexpr size_t OFF_CNT    = OFF_TEW2 + (size_t)kE * kH * kO * 2;  // int [kE]
constexpr size_t OFF_POFF   = OFF_CNT + 256;                        // int [kE+1]
constexpr size_t OFF_LIST   = OFF_POFF + 256;                       // int [kE][kT]
constexpr size_t OFF_WT     = OFF_LIST + (size_t)kE * kT * 4;       // float [kE][kT]

// ---------------- types ----------------
typedef __attribute__((ext_vector_type(16))) __bf16 v16bf;
typedef __attribute__((ext_vector_type(8)))  __bf16 v8bf;
typedef __attribute__((ext_vector_type(8)))  float  v8f;
typedef __attribute__((ext_vector_type(4)))  unsigned int v4u;
typedef __attribute__((ext_vector_type(8)))  int v8i_;
typedef __attribute__((ext_vector_type(4)))  int v4i_;

// Tensor Data Mover availability (arity differs across toolchains)
#if defined(__has_builtin)
#if __has_builtin(__builtin_amdgcn_tensor_load_to_lds) && \
    __has_builtin(__builtin_amdgcn_s_wait_tensorcnt)
#define USE_TDM 1
#endif
#endif
#ifndef USE_TDM
#define USE_TDM 0
#endif

// ---- WMMA fragment loads from LDS, per ISA 7.12.2 16-bit layouts ----
__device__ __forceinline__ v16bf frag_a(const __bf16* ldsA, int m0, int lane) {
  const __bf16* p = ldsA + (m0 + (lane & 15)) * BK + ((lane >> 4) << 3);
  v8bf lo = *(const v8bf*)p;          // K base+0..7
  v8bf hi = *(const v8bf*)(p + 16);   // K base+16..23
  v16bf a;
#pragma unroll
  for (int i = 0; i < 8; ++i) { a[i] = lo[i]; a[i + 8] = hi[i]; }
  return a;
}
__device__ __forceinline__ v16bf frag_b(const __bf16* ldsB, int n0, int lane) {
  const __bf16* p = ldsB + (n0 + (lane & 15)) * BK + ((lane >> 4) << 3);
  v8bf lo = *(const v8bf*)p;
  v8bf hi = *(const v8bf*)(p + 16);
  v16bf b;
#pragma unroll
  for (int i = 0; i < 8; ++i) { b[i] = lo[i]; b[i + 8] = hi[i]; }
  return b;
}
__device__ __forceinline__ v8f wmma_bf16(v16bf a, v16bf b, v8f c) {
  return __builtin_amdgcn_wmma_f32_16x16x32_bf16(
      false, a, false, b, (short)0, c, false, false);
}

// ---------------- kernel 0: fp32 [K][N] -> bf16 [N][K] tiled transpose-cast ----
__global__ __launch_bounds__(256) void k_transpose_cast(
    const float* __restrict__ src, __bf16* __restrict__ dst, int K, int N) {
  __shared__ float tile[32][33];
  const size_t bo = (size_t)blockIdx.z * K * N;
  const int kb = blockIdx.y * 32, nb = blockIdx.x * 32;
  const int c = threadIdx.x & 31, r0 = threadIdx.x >> 5;
#pragma unroll
  for (int rr = 0; rr < 32; rr += 8)
    tile[r0 + rr][c] = src[bo + (size_t)(kb + r0 + rr) * N + nb + c];
  __syncthreads();
#pragma unroll
  for (int rr = 0; rr < 32; rr += 8)
    dst[bo + (size_t)(nb + r0 + rr) * K + kb + c] = (__bf16)tile[c][r0 + rr];
}

// ---------------- kernel 1: RMSNorm (eps=1e-8) + cast to bf16 ----------------
__global__ __launch_bounds__(256) void k_rmsnorm(const float* __restrict__ x,
                                                 const float* __restrict__ g,
                                                 __bf16* __restrict__ out) {
  const int t = blockIdx.x, tid = threadIdx.x;
  __shared__ float red[8];
  __shared__ float rmsv;
  const float* row = x + (size_t)t * kD;
  float v[4]; float s = 0.f;
#pragma unroll
  for (int i = 0; i < 4; ++i) { v[i] = row[tid + 256 * i]; s += v[i] * v[i]; }
#pragma unroll
  for (int off = 16; off; off >>= 1) s += __shfl_xor(s, off, 32);
  if ((tid & 31) == 0) red[tid >> 5] = s;
  __syncthreads();
  if (tid == 0) {
    float a = 0.f;
#pragma unroll
    for (int i = 0; i < 8; ++i) a += red[i];
    rmsv = rsqrtf(a / (float)kD + 1e-8f);
  }
  __syncthreads();
  const float r = rmsv;
#pragma unroll
  for (int i = 0; i < 4; ++i)
    out[(size_t)t * kD + tid + 256 * i] = (__bf16)(v[i] * r * g[tid + 256 * i]);
}

// ---------------- kernel 2: sigmoid router, top-2, scatter lists ----------------
__global__ __launch_bounds__(256) void k_router(const __bf16* __restrict__ normed,
                                                const float* __restrict__ rw,
                                                int* __restrict__ cnt,
                                                int* __restrict__ list,
                                                float* __restrict__ wt) {
  const int lane = threadIdx.x & 31, wid = threadIdx.x >> 5;
  const int t = blockIdx.x * 8 + wid;
  const __bf16* a = normed + (size_t)t * kD;
  float logit[kE];
#pragma unroll
  for (int e = 0; e < kE; ++e) {
    const float* w = rw + (size_t)e * kD;
    float s = 0.f;
    for (int j = lane; j < kD; j += 32) s += (float)a[j] * w[j];
#pragma unroll
    for (int off = 16; off; off >>= 1) s += __shfl_xor(s, off, 32);
    logit[e] = s;
  }
  if (lane == 0) {
    float p[kE];
#pragma unroll
    for (int e = 0; e < kE; ++e) p[e] = 1.f / (1.f + __expf(-logit[e]));
    int i0 = 0;
#pragma unroll
    for (int e = 1; e < kE; ++e) if (p[e] > p[i0]) i0 = e;
    int i1 = (i0 == 0) ? 1 : 0;
#pragma unroll
    for (int e = 0; e < kE; ++e) if (e != i0 && p[e] > p[i1]) i1 = e;
    const float inv = 1.f / (p[i0] + p[i1] + 1e-6f);
    int s0 = atomicAdd(&cnt[i0], 1);
    list[i0 * kT + s0] = t; wt[i0 * kT + s0] = p[i0] * inv;
    int s1 = atomicAdd(&cnt[i1], 1);
    list[i1 * kT + s1] = t; wt[i1 * kT + s1] = p[i1] * inv;
  }
}

// ---------------- kernel 3: 128-padded prefix offsets per expert ----------------
__global__ void k_scan(const int* __restrict__ cnt, int* __restrict__ poff) {
  if (threadIdx.x == 0) {
    int acc = 0; poff[0] = 0;
    for (int e = 0; e < kE; ++e) {
      acc += ((cnt[e] + BM - 1) / BM) * BM;
      poff[e + 1] = acc;
    }
  }
}

// ---------------- kernel 4: bf16 WMMA GEMM, 128x128 tile, K-step 32 ----------
// Double-buffered software pipeline: B tile DMA'd by the Tensor Data Mover into
// buf[i^1] while WMMAs consume buf[i]; A tile vector-filled in the same shadow.
// MODE 0: shared FC1  (A=normed direct,   out=bf16 hsh, +bias, ReLU)
// MODE 1: expert FC1  (A=normed gathered, out=bf16 eh,  +bias, ReLU)
// MODE 2: shared FC2  (A=hsh direct,      out=f32 d_out, +bias)
// MODE 3: expert FC2  (A=eh direct,       weighted atomicAdd into d_out, +bias)
template <int MODE, int KD, int NT>
__global__ __launch_bounds__(256) void k_gemm(
    const __bf16* __restrict__ Abase, const __bf16* __restrict__ Wt,
    const float* __restrict__ biasBase, __bf16* __restrict__ bfOut,
    float* __restrict__ fOut, const int* __restrict__ cnt,
    const int* __restrict__ poff, const int* __restrict__ list,
    const float* __restrict__ wts) {
  __shared__ __bf16 lA[2][BM * BK];   // 2 x 8 KB
  __shared__ __bf16 lB[2][BN * BK];   // 2 x 8 KB, [N][K]
  constexpr int NSTEPS = KD / BK;

  int e = 0, tile, cE = 0, pE = 0;
  if (MODE == 1 || MODE == 3) {
    e = blockIdx.y >> 5;          // 32 tiles per expert
    tile = blockIdx.y & 31;
    cE = cnt[e];
    if (tile * BM >= cE) return;  // uniform across block
    pE = poff[e];
  } else {
    tile = blockIdx.y;
  }
  const int n0 = blockIdx.x * BN;
  const int tid = threadIdx.x, lane = tid & 31, wid = tid >> 5;
  const int wm = (wid & 3) * 32;   // wave: 32 rows x 64 cols
  const int wn = (wid >> 2) * 64;

  const __bf16* W = Wt + (size_t)((MODE == 1 || MODE == 3) ? e : 0) * KD * NT;
  const float* bias = biasBase + ((MODE == 1 || MODE == 3) ? e * NT : 0);

  // per-thread fill coords: row = tid>>1, 16-element segment = tid&1
  const int ra = tid >> 1, seg = tid & 1;
  const __bf16* aptr;           // A source row (nullptr => zero row, MODE 1 only)
  if (MODE == 1) {
    int idx = tile * BM + ra;
    aptr = (idx < cE) ? Abase + (size_t)list[e * kT + idx] * KD : nullptr;
  } else if (MODE == 3) {
    aptr = Abase + (size_t)(pE + tile * BM + ra) * KD;
  } else {
    aptr = Abase + (size_t)(tile * BM + ra) * KD;
  }

  auto fillA = [&](int kk, int b) {
    uint4* dst = (uint4*)&lA[b][ra * BK + seg * 16];
    if (MODE != 1) {
      const uint4* s = (const uint4*)(aptr + kk + seg * 16);
      dst[0] = s[0]; dst[1] = s[1];
    } else if (aptr) {
      const uint4* s = (const uint4*)(aptr + kk + seg * 16);
      dst[0] = s[0]; dst[1] = s[1];
    } else {
      dst[0] = make_uint4(0u, 0u, 0u, 0u);
      dst[1] = make_uint4(0u, 0u, 0u, 0u);
    }
  };

#if USE_TDM
  // Tensor DMA descriptor (ISA 8.3/8.4): 2D tile 32(K) x 128(N), 2-byte elems
  const unsigned long long wga0 =
      (unsigned long long)(size_t)(const void*)(W + (size_t)n0 * KD);
  const unsigned ldsBoff[2] = { (unsigned)(size_t)(void*)&lB[0][0],
                                (unsigned)(size_t)(void*)&lB[1][0] };
  const v4i_ z4 = {0, 0, 0, 0};
#if __clang_major__ >= 23
  const v8i_ z8 = {0, 0, 0, 0, 0, 0, 0, 0};
#endif
  auto issueB = [&](int kk, int b) {
    if (wid == 0) {  // wave 0 issues the weight-tile DMA (EXEC ignored by TDM)
      unsigned long long ga = wga0 + (unsigned long long)kk * 2ull;
      v4u g0 = { 1u,                                   // count=1, user descriptor
                 ldsBoff[b],                            // lds_addr
                 (unsigned)ga,                          // global_addr[31:0]
                 (unsigned)(ga >> 32) | (2u << 30) };   // global_addr[56:32] | type=2
      v8i_ g1 = { (int)0x00010000u,                               // data_size=2B
                  (int)(((unsigned)KD & 0xFFFFu) << 16),          // tensor_dim0 lo16
                  (int)(((unsigned)KD >> 16) |
                        (((unsigned)NT & 0xFFFFu) << 16)),        // tdim0 hi | tdim1 lo
                  (int)(((unsigned)NT >> 16) | (32u << 16)),      // tdim1 hi | tile_dim0
                  (int)128,                                       // tile_dim1=128
                  (int)KD,                                        // tensor_dim0_stride
                  0, 0 };
#if __clang_major__ >= 23
      __builtin_amdgcn_tensor_load_to_lds(g0, g1, z4, z4, z8, 0);
#else
      __builtin_amdgcn_tensor_load_to_lds(g0, g1, z4, z4, 0);
#endif
    }
  };
#else
  const __bf16* bptr = W + (size_t)(n0 + ra) * KD;
  auto issueB = [&](int kk, int b) {
    uint4* dst = (uint4*)&lB[b][ra * BK + seg * 16];
    const uint4* s = (const uint4*)(bptr + kk + seg * 16);
    dst[0] = s[0]; dst[1] = s[1];
  };
#endif

  v8f acc[2][4] = {};

  // prologue: stage step 0 into buffer 0
  issueB(0, 0);
  fillA(0, 0);

  for (int i = 0; i < NSTEPS; ++i) {
    const int cur = i & 1;
#if USE_TDM
    // wait for DMA into buf[cur] (no-op on waves that issued nothing)
    __builtin_amdgcn_s_wait_tensorcnt(0);
#endif
    __syncthreads();   // buf[cur] ready for all; buf[cur^1] reads from i-1 done
    if (i + 1 < NSTEPS) {          // overlap next tile's DMA + A loads w/ compute
      issueB((i + 1) * BK, cur ^ 1);
      fillA((i + 1) * BK, cur ^ 1);
    }
    v16bf a0 = frag_a(lA[cur], wm, lane);
    v16bf a1 = frag_a(lA[cur], wm + 16, lane);
#pragma unroll
    for (int j = 0; j < 4; ++j) {
      v16bf b = frag_b(lB[cur], wn + j * 16, lane);
      acc[0][j] = wmma_bf16(a0, b, acc[0][j]);
      acc[1][j] = wmma_bf16(a1, b, acc[1][j]);
    }
  }

  // ---- epilogue ----
  float bv[4];
#pragma unroll
  for (int j = 0; j < 4; ++j) bv[j] = bias[n0 + wn + j * 16 + (lane & 15)];
#pragma unroll
  for (int im = 0; im < 2; ++im) {
#pragma unroll
    for (int r = 0; r < 8; ++r) {
      const int mloc = wm + im * 16 + (lane >> 4) * 8 + r;
      int tok = -1; float wgt = 0.f; size_t orow = 0;
      if (MODE == 0 || MODE == 2) orow = (size_t)(tile * BM + mloc);
      if (MODE == 1)              orow = (size_t)(pE + tile * BM + mloc);
      if (MODE == 3) {
        int idx = tile * BM + mloc;
        if (idx < cE) { tok = list[e * kT + idx]; wgt = wts[e * kT + idx]; }
      }
#pragma unroll
      for (int j = 0; j < 4; ++j) {
        const int n = n0 + wn + j * 16 + (lane & 15);
        float v = acc[im][j][r] + bv[j];
        if (MODE == 0 || MODE == 1) {
          v = v > 0.f ? v : 0.f;                       // ReLU
          bfOut[orow * NT + n] = (__bf16)v;
        } else if (MODE == 2) {
          fOut[orow * NT + n] = v;                     // initializes d_out
        } else {
          if (tok >= 0) atomicAdd(&fOut[(size_t)tok * NT + n], wgt * v);
        }
      }
    }
  }
}

// ---------------- host ----------------
extern "C" void kernel_launch(void* const* d_in, const int* in_sizes, int n_in,
                              void* d_out, int out_size, void* d_ws, size_t ws_size,
                              hipStream_t stream) {
  const float* hs  = (const float*)d_in[0];
  const float* nw  = (const float*)d_in[1];
  const float* rw  = (const float*)d_in[2];
  const float* sw1 = (const float*)d_in[3];
  const float* sb1 = (const float*)d_in[4];
  const float* sw2 = (const float*)d_in[5];
  const float* sb2 = (const float*)d_in[6];
  const float* ew1 = (const float*)d_in[7];
  const float* eb1 = (const float*)d_in[8];
  const float* ew2 = (const float*)d_in[9];
  const float* eb2 = (const float*)d_in[10];
  float* out = (float*)d_out;

  char* ws = (char*)d_ws;
  __bf16* normed = (__bf16*)(ws + OFF_NORMED);
  __bf16* hsh    = (__bf16*)(ws + OFF_HSH);
  __bf16* eh     = (__bf16*)(ws + OFF_EH);
  __bf16* tsw1   = (__bf16*)(ws + OFF_TSW1);
  __bf16* tsw2   = (__bf16*)(ws + OFF_TSW2);
  __bf16* tew1   = (__bf16*)(ws + OFF_TEW1);
  __bf16* tew2   = (__bf16*)(ws + OFF_TEW2);
  int*   cnt  = (int*)(ws + OFF_CNT);
  int*   poff = (int*)(ws + OFF_POFF);
  int*   list = (int*)(ws + OFF_LIST);
  float* wt   = (float*)(ws + OFF_WT);

  hipMemsetAsync(cnt, 0, 256, stream);

  // weights: fp32 [K][N] -> bf16 [N][K] once per launch (~430 MB traffic, ~19us);
  // bf16 expert weights (128 MB) then fit in the 192 MB L2 across row-tile passes.
  k_transpose_cast<<<dim3(kH / 32, kD / 32, 1),  256, 0, stream>>>(sw1, tsw1, kD, kH);
  k_transpose_cast<<<dim3(kO / 32, kH / 32, 1),  256, 0, stream>>>(sw2, tsw2, kH, kO);
  k_transpose_cast<<<dim3(kH / 32, kD / 32, kE), 256, 0, stream>>>(ew1, tew1, kD, kH);
  k_transpose_cast<<<dim3(kO / 32, kH / 32, kE), 256, 0, stream>>>(ew2, tew2, kH, kO);

  k_rmsnorm<<<kT, 256, 0, stream>>>(hs, nw, normed);
  k_router<<<kT / 8, 256, 0, stream>>>(normed, rw, cnt, list, wt);
  k_scan<<<1, 32, 0, stream>>>(cnt, poff);

  // shared expert: FC1 (T x D x H) then FC2 (T x H x O) -> writes d_out
  k_gemm<0, kD, kH><<<dim3(kH / BN, kT / BM), 256, 0, stream>>>(
      normed, tsw1, sb1, hsh, out, cnt, poff, list, wt);
  k_gemm<2, kH, kO><<<dim3(kO / BN, kT / BM), 256, 0, stream>>>(
      hsh, tsw2, sb2, hsh, out, cnt, poff, list, wt);

  // routed experts (top-2 gather): FC1 then weighted scatter FC2 -> atomicAdd d_out
  k_gemm<1, kD, kH><<<dim3(kH / BN, kE * TPE), 256, 0, stream>>>(
      normed, tew1, eb1, eh, out, cnt, poff, list, wt);
  k_gemm<3, kH, kO><<<dim3(kO / BN, kE * TPE), 256, 0, stream>>>(
      eh, tew2, eb2, eh, out, cnt, poff, list, wt);
}
